// Injector_60464549593389
// MI455X (gfx1250) — matrix operand order repaired and checked by reading, not compiled, not run on hardware
//
#include <hip/hip_runtime.h>
#include <hip/hip_bf16.h>

// ---------------------------------------------------------------------------
// MSDA (deformable attention) for MI455X / gfx1250.
// GEMMs: v_wmma_f32_16x16x32_bf16, fp32 accumulate.
// Weight tiles staged to LDS with the Tensor Data Mover (TENSOR_LOAD_TO_LDS),
// double-buffered on TENSORcnt; TDM OOB-returns-zero handles the N tail.
// ---------------------------------------------------------------------------

typedef __attribute__((ext_vector_type(16))) __bf16 v16bf;
typedef __attribute__((ext_vector_type(8)))  __bf16 v8bf;
typedef __attribute__((ext_vector_type(4)))  __bf16 v4bf;
typedef __attribute__((ext_vector_type(8)))  float  v8f;
typedef __attribute__((ext_vector_type(4)))  unsigned int u32x4;
typedef __attribute__((ext_vector_type(8)))  int          i32x8;
typedef __attribute__((ext_vector_type(4)))  int          i32x4;

#define B_   2
#define NQ_  4096
#define C_   1024
#define NH_  8
#define NL_  3
#define NP_  4
#define NV_  21504   // 128*128 + 64*64 + 32*32

#define KC_     128                 // K elements staged per TDM chunk
#define BROWS_  64                  // N rows per block tile
#define PITCH_  (KC_ * 2 + 16)      // 272B: TDM pad keeps 16B align, de-conflicts banks

union ABf { v16bf v; v8bf h[2]; };

// ---------------------------------------------------------------------------
// TDM: stage Wt[nBase .. nBase+64) x [kb .. kb+KC_) (bf16) into LDS.
// D# per CDNA5 ISA sec.8. OOB rows (>= N-nBase) read zero.
// Padding: after every 256B (pad_interval code 5) insert 16B (pad_amount code 3).
// ---------------------------------------------------------------------------
static __device__ inline void tdm_load_b_tile(const __bf16* Wt, int N, int K,
                                              int nBase, int kb, unsigned ldsAddr) {
    const unsigned long long gaddr =
        (unsigned long long)(const void*)Wt +
        ((unsigned long long)nBase * (unsigned long long)K + (unsigned long long)kb) * 2ull;
    const unsigned dim1 = (unsigned)(N - nBase);     // remaining valid rows

    u32x4 g0;
    g0[0] = 1u;                                             // count=1, no gather
    g0[1] = ldsAddr;                                        // LDS byte address
    g0[2] = (unsigned)(gaddr & 0xFFFFFFFFull);              // global_addr[31:0]
    g0[3] = (unsigned)((gaddr >> 32) & 0x01FFFFFFull)       // global_addr[56:32]
          | (2u << 30);                                     // type=2 ("image")

    i32x8 g1;
    g1[0] = (int)((1u << 16)        // data_size = 1 -> 2 bytes
                | (1u << 20)        // pad_enable
                | (5u << 22)        // pad_interval: 64 dwords = 256B
                | (3u << 25));      // pad_amount:   4 dwords = 16B
    g1[1] = (int)(((unsigned)K & 0xFFFFu) << 16);           // tensor_dim0[15:0]
    g1[2] = (int)((dim1 & 0xFFFFu) << 16);                  // tensor_dim1[15:0]
    g1[3] = (int)((unsigned)KC_ << 16);                     // tile_dim0
    g1[4] = BROWS_;                                         // tile_dim1 (tile_dim2=0)
    g1[5] = K;                                              // tensor_dim0_stride[31:0]
    g1[6] = 0;
    g1[7] = 0;

    i32x4 z4; z4[0] = 0; z4[1] = 0; z4[2] = 0; z4[3] = 0;
#if defined(__clang_major__) && (__clang_major__ >= 23)
    i32x8 z8;
#pragma unroll
    for (int i = 0; i < 8; ++i) z8[i] = 0;
    __builtin_amdgcn_tensor_load_to_lds(g0, g1, z4, z4, z8, 0);
#else
    __builtin_amdgcn_tensor_load_to_lds(g0, g1, z4, z4, 0);
#endif
}

// ---------------------------------------------------------------------------
// LayerNorm fp32 -> bf16. One 256-thread block per row of 1024.
// ---------------------------------------------------------------------------
__global__ __launch_bounds__(256)
void layernorm_bf16(const float* __restrict__ x, const float* __restrict__ w,
                    const float* __restrict__ bsh, __bf16* __restrict__ out) {
    __shared__ float s1[256];
    __shared__ float s2[256];
    const int row = blockIdx.x;
    const int tid = threadIdx.x;
    const float4 v = *(const float4*)(x + (size_t)row * C_ + tid * 4);
    s1[tid] = v.x + v.y + v.z + v.w;
    s2[tid] = v.x * v.x + v.y * v.y + v.z * v.z + v.w * v.w;
    __syncthreads();
#pragma unroll
    for (int o = 128; o > 0; o >>= 1) {
        if (tid < o) { s1[tid] += s1[tid + o]; s2[tid] += s2[tid + o]; }
        __syncthreads();
    }
    const float mu  = s1[0] * (1.0f / C_);
    const float var = s2[0] * (1.0f / C_) - mu * mu;
    const float ri  = rsqrtf(var + 1e-6f);
    const int c = tid * 4;
    v4bf o;
    o[0] = (__bf16)((v.x - mu) * ri * w[c + 0] + bsh[c + 0]);
    o[1] = (__bf16)((v.y - mu) * ri * w[c + 1] + bsh[c + 1]);
    o[2] = (__bf16)((v.z - mu) * ri * w[c + 2] + bsh[c + 2]);
    o[3] = (__bf16)((v.w - mu) * ri * w[c + 3] + bsh[c + 3]);
    *(v4bf*)(out + (size_t)row * C_ + c) = o;
}

// ---------------------------------------------------------------------------
// Weight transpose + cast: W fp32 [K,N] -> Wt bf16 [N,K]
// ---------------------------------------------------------------------------
__global__ __launch_bounds__(256)
void transpose_bf16(const float* __restrict__ W, __bf16* __restrict__ Wt,
                    int K, int N) {
    const int idx = blockIdx.x * 256 + threadIdx.x;
    if (idx >= K * N) return;
    const int k = idx / N;
    const int n = idx - k * N;
    Wt[(size_t)n * K + k] = (__bf16)W[idx];
}

// ---------------------------------------------------------------------------
// WMMA GEMM: out[M,N] = A[M,K](bf16 row-major) @ Wt[N,K]^T + bias
// block = 256 thr (8 waves); wave tile = 32(M) x 64(N); K chunked by TDM.
// MODE 0: fp32 out   MODE 1: bf16 out   MODE 2: fp32 out = resid + gamma*(.)
// ---------------------------------------------------------------------------
template<int MODE>
__global__ __launch_bounds__(256, 2)
void gemm_bf16_wmma(const __bf16* __restrict__ A, const __bf16* __restrict__ Wt,
                    const float* __restrict__ bias, void* __restrict__ outp,
                    const float* __restrict__ resid, const float* __restrict__ gamma,
                    int N, int K) {
    __shared__ __align__(16) char smem[2][BROWS_ * PITCH_];

    const int lane   = threadIdx.x & 31;
    const int wave   = threadIdx.x >> 5;
    const int laneHi = (lane >> 4) & 1;
    const int l16    = lane & 15;
    const int mBase  = blockIdx.y * 256 + wave * 32;
    const int nBase  = blockIdx.x * 64;

    v8f acc[2][4] = {};

    const int aOff = laneHi ? 8 : 0;    // A frag: lo lanes K+0..7/16..23, hi +8
    const int bOff = laneHi ? 8 : 0;    // B frag: lo lanes K+0..7/16..23, hi +8
    const __bf16* a0p = A + (size_t)(mBase + l16) * K + aOff;
    const __bf16* a1p = a0p + (size_t)16 * K;

    // Prologue: wave 0 DMAs chunk 0 into buffer 0.
    if (wave == 0) {
        tdm_load_b_tile(Wt, N, K, nBase, 0, (unsigned)(size_t)&smem[0][0]);
    }

    const int nChunks = K / KC_;
    for (int ci = 0; ci < nChunks; ++ci) {
        const int kb = ci * KC_;
        __syncthreads();   // everyone done reading buffer (ci+1)&1 from chunk ci-1
        if (wave == 0) {
            if (ci + 1 < nChunks) {
                tdm_load_b_tile(Wt, N, K, nBase, kb + KC_,
                                (unsigned)(size_t)&smem[(ci + 1) & 1][0]);
                __builtin_amdgcn_s_wait_tensorcnt(1);   // chunk ci complete
            } else {
                __builtin_amdgcn_s_wait_tensorcnt(0);
            }
        }
        __syncthreads();   // chunk ci visible to all waves
        const char* bbase = &smem[ci & 1][0];

#pragma unroll
        for (int k0 = 0; k0 < KC_; k0 += 32) {
            ABf a0, a1;
            a0.h[0] = *(const v8bf*)(a0p + kb + k0);
            a0.h[1] = *(const v8bf*)(a0p + kb + k0 + 16);
            a1.h[0] = *(const v8bf*)(a1p + kb + k0);
            a1.h[1] = *(const v8bf*)(a1p + kb + k0 + 16);
#pragma unroll
            for (int ni = 0; ni < 4; ++ni) {
                const char* bp = bbase + (size_t)(ni * 16 + l16) * PITCH_
                                       + (size_t)(k0 + bOff) * 2;
                ABf b;
                b.h[0] = *(const v8bf*)(bp);
                b.h[1] = *(const v8bf*)(bp + 32);
                acc[0][ni] = __builtin_amdgcn_wmma_f32_16x16x32_bf16(
                    false, a0.v, false, b.v, (short)0, acc[0][ni], false, false);
                acc[1][ni] = __builtin_amdgcn_wmma_f32_16x16x32_bf16(
                    false, a1.v, false, b.v, (short)0, acc[1][ni], false, false);
            }
        }
    }

    // Epilogue: D layout = VGPR r, lanes 0-15 -> M=r, lanes 16-31 -> M=r+8.
#pragma unroll
    for (int mi = 0; mi < 2; ++mi) {
#pragma unroll
        for (int ni = 0; ni < 4; ++ni) {
            if (nBase + ni * 16 >= N) continue;          // wave-uniform tail guard
            const int ncol = nBase + ni * 16 + l16;
            const float bb = bias[ncol];
#pragma unroll
            for (int r = 0; r < 8; ++r) {
                const int row  = mBase + mi * 16 + r + (laneHi ? 8 : 0);
                const size_t id = (size_t)row * N + ncol;
                const float v = acc[mi][ni][r] + bb;
                if (MODE == 0) {
                    ((float*)outp)[id] = v;
                } else if (MODE == 1) {
                    ((__bf16*)outp)[id] = (__bf16)v;
                } else {
                    ((float*)outp)[id] = resid[id] + gamma[ncol] * v;
                }
            }
        }
    }
}

// ---------------------------------------------------------------------------
// Softmax over the 12 (level,point) logits per (b,q,head). In-place.
// ---------------------------------------------------------------------------
__global__ __launch_bounds__(256)
void softmax12(float* __restrict__ aw) {
    const int r = blockIdx.x * 256 + threadIdx.x;   // < B*NQ*NH = 65536
    float* p = aw + (size_t)r * 12;
    float v[12];
    float m = -1e30f;
#pragma unroll
    for (int i = 0; i < 12; ++i) { v[i] = p[i]; m = fmaxf(m, v[i]); }
    float s = 0.0f;
#pragma unroll
    for (int i = 0; i < 12; ++i) { v[i] = __expf(v[i] - m); s += v[i]; }
    const float inv = 1.0f / s;
#pragma unroll
    for (int i = 0; i < 12; ++i) p[i] = v[i] * inv;
}

// ---------------------------------------------------------------------------
// Bilinear sampling + attention-weighted aggregation.
// grid = (NQ, B); block = 256 = 8 waves; wave h = head h; lane owns 4 channels
// (8B coalesced gathers from L2-resident bf16 value tensor).
// All tap branches are wave-uniform -> EXEC stays full.
// ---------------------------------------------------------------------------
__global__ __launch_bounds__(256)
void msda_sample(const __bf16* __restrict__ value, const float* __restrict__ off,
                 const float* __restrict__ aw, const float* __restrict__ refp,
                 __bf16* __restrict__ attn) {
    const int q    = blockIdx.x;
    const int b    = blockIdx.y;
    const int h    = threadIdx.x >> 5;
    const int lane = threadIdx.x & 31;
    const int c    = lane * 4;
    const size_t bq = (size_t)b * NQ_ + q;

    const float* op = off + (bq * NH_ + h) * (NL_ * NP_ * 2);  // [h][l][p][2]
    const float* ap = aw  + (bq * NH_ + h) * (NL_ * NP_);      // [h][l*4+p]

    float acc0 = 0.f, acc1 = 0.f, acc2 = 0.f, acc3 = 0.f;

    const int Hs[3] = {128, 64, 32};
    const int Wd[3] = {128, 64, 32};
    const int st[3] = {0, 16384, 20480};

#pragma unroll
    for (int l = 0; l < NL_; ++l) {
        const float rx = refp[(bq * NL_ + l) * 2 + 0];
        const float ry = refp[(bq * NL_ + l) * 2 + 1];
        const int Hc = Hs[l], Wc = Wd[l], start = st[l];
        const float fW = (float)Wc, fH = (float)Hc;
#pragma unroll
        for (int p = 0; p < NP_; ++p) {
            const float ox = op[(l * 4 + p) * 2 + 0];
            const float oy = op[(l * 4 + p) * 2 + 1];
            const float px = (rx + ox / fW) * fW - 0.5f;
            const float py = (ry + oy / fH) * fH - 0.5f;
            const float x0f = floorf(px), y0f = floorf(py);
            const float fx = px - x0f, fy = py - y0f;
            const int x0 = (int)x0f, y0 = (int)y0f;
            const float wA = ap[l * 4 + p];
            const float tw[4] = { (1.f - fx) * (1.f - fy), fx * (1.f - fy),
                                  (1.f - fx) * fy,          fx * fy };
            const int xs[4] = { x0, x0 + 1, x0,     x0 + 1 };
            const int ys[4] = { y0, y0,     y0 + 1, y0 + 1 };
#pragma unroll
            for (int t = 0; t < 4; ++t) {
                const int xi = xs[t], yi = ys[t];
                if (xi >= 0 && xi < Wc && yi >= 0 && yi < Hc) {   // wave-uniform
                    const size_t vid = ((size_t)b * NV_ + start + (size_t)yi * Wc + xi)
                                       * (size_t)C_ + (size_t)h * 128 + c;
                    const v4bf vv = *(const v4bf*)(value + vid);
                    const float wt = wA * tw[t];
                    acc0 += wt * (float)vv[0];
                    acc1 += wt * (float)vv[1];
                    acc2 += wt * (float)vv[2];
                    acc3 += wt * (float)vv[3];
                }
            }
        }
    }
    v4bf o;
    o[0] = (__bf16)acc0; o[1] = (__bf16)acc1;
    o[2] = (__bf16)acc2; o[3] = (__bf16)acc3;
    *(v4bf*)(attn + bq * C_ + (size_t)h * 128 + c) = o;
}

// ---------------------------------------------------------------------------
// Host launcher
// ---------------------------------------------------------------------------
extern "C" void kernel_launch(void* const* d_in, const int* in_sizes, int n_in,
                              void* d_out, int out_size, void* d_ws, size_t ws_size,
                              hipStream_t stream) {
    const float* query  = (const float*)d_in[0];
    const float* refp   = (const float*)d_in[1];
    const float* feat   = (const float*)d_in[2];
    const float* q_ln_w = (const float*)d_in[3];
    const float* q_ln_b = (const float*)d_in[4];
    const float* f_ln_w = (const float*)d_in[5];
    const float* f_ln_b = (const float*)d_in[6];
    const float* Wv     = (const float*)d_in[7];
    const float* bv     = (const float*)d_in[8];
    const float* Ws     = (const float*)d_in[9];
    const float* bs     = (const float*)d_in[10];
    const float* Wa     = (const float*)d_in[11];
    const float* ba     = (const float*)d_in[12];
    const float* Wo     = (const float*)d_in[13];
    const float* bo     = (const float*)d_in[14];
    const float* gamma  = (const float*)d_in[15];
    // d_in[16] spatial_shapes, d_in[17] lvl_start: static, hardcoded.

    const size_t RQ = (size_t)B_ * NQ_;   // 8192
    const size_t RV = (size_t)B_ * NV_;   // 43008

    char* ws = (char*)d_ws;
    __bf16* q_ln  = (__bf16*)ws;  ws += RQ * C_ * 2;
    __bf16* v_ln  = (__bf16*)ws;  ws += RV * C_ * 2;
    __bf16* Wvt   = (__bf16*)ws;  ws += (size_t)C_ * C_ * 2;
    __bf16* Wst   = (__bf16*)ws;  ws += (size_t)192 * C_ * 2;
    __bf16* Wat   = (__bf16*)ws;  ws += (size_t)96 * C_ * 2;
    __bf16* Wot   = (__bf16*)ws;  ws += (size_t)C_ * C_ * 2;
    __bf16* value = (__bf16*)ws;  ws += RV * C_ * 2;
    float*  offb  = (float*)ws;   ws += RQ * 192 * 4;
    float*  awb   = (float*)ws;   ws += RQ * 96 * 4;
    __bf16* attn  = (__bf16*)ws;  ws += RQ * C_ * 2;

    const dim3 blk(256);

    // 1) LayerNorms -> bf16
    layernorm_bf16<<<dim3((unsigned)RQ), blk, 0, stream>>>(query, q_ln_w, q_ln_b, q_ln);
    layernorm_bf16<<<dim3((unsigned)RV), blk, 0, stream>>>(feat,  f_ln_w, f_ln_b, v_ln);

    // 2) Weight transposes -> bf16 [N,K]
    transpose_bf16<<<dim3((C_ * C_)  / 256), blk, 0, stream>>>(Wv, Wvt, C_, C_);
    transpose_bf16<<<dim3((C_ * 192) / 256), blk, 0, stream>>>(Ws, Wst, C_, 192);
    transpose_bf16<<<dim3((C_ * 96)  / 256), blk, 0, stream>>>(Wa, Wat, C_, 96);
    transpose_bf16<<<dim3((C_ * C_)  / 256), blk, 0, stream>>>(Wo, Wot, C_, C_);

    // 3) Value projection (bf16 out; feeds L2-resident gather)
    gemm_bf16_wmma<1><<<dim3(C_ / 64, (unsigned)(RV / 256)), blk, 0, stream>>>(
        v_ln, Wvt, bv, value, nullptr, nullptr, C_, C_);

    // 4) Sampling offsets (N=192) and attention logits (N=96, TDM zero-fills tail)
    gemm_bf16_wmma<0><<<dim3(3, (unsigned)(RQ / 256)), blk, 0, stream>>>(
        q_ln, Wst, bs, offb, nullptr, nullptr, 192, C_);
    gemm_bf16_wmma<0><<<dim3(2, (unsigned)(RQ / 256)), blk, 0, stream>>>(
        q_ln, Wat, ba, awb, nullptr, nullptr, 96, C_);

    // 5) Softmax over 12 slots per (b,q,h)
    softmax12<<<dim3((unsigned)(RQ * NH_ / 256)), blk, 0, stream>>>(awb);

    // 6) Bilinear sampling + aggregation -> attn (bf16)
    msda_sample<<<dim3(NQ_, B_), blk, 0, stream>>>(value, offb, awb, refp, attn);

    // 7) Output projection + residual: out = query + gamma * (attn @ Wo + bo)
    gemm_bf16_wmma<2><<<dim3(C_ / 64, (unsigned)(RQ / 256)), blk, 0, stream>>>(
        attn, Wot, bo, d_out, query, gamma, C_, C_);
}
